// Attention_88313117540497
// MI455X (gfx1250) — matrix operand order repaired and checked by reading, not compiled, not run on hardware
//
#include <hip/hip_runtime.h>
#include <hip/hip_bf16.h>

// ---------------------------------------------------------------------------
// Fused attention block on MI455X (gfx1250):
//   v   = x @ Wv                  (25088x384 @ 384x512)
//   mid = M @ v   per batch       (196x196  @ 196x512, x128 batches)
//         where M[p,q] = qk[13+qy-py, 13+qx-px]  (conv == position mixing)
//   out = mid @ Wo + bo           (25088x512 @ 512x384)
// All GEMMs run as f16-operand WMMA (v_wmma_f32_16x16x32_f16) with f32
// accumulation. Intermediates live in d_ws as f16 (~49 MB, L2-resident).
// Double-buffered LDS staging; f16 A-tiles use GLOBAL_LOAD_ASYNC_TO_LDS_B64.
// ---------------------------------------------------------------------------

typedef __attribute__((ext_vector_type(16))) _Float16     v16h;
typedef __attribute__((ext_vector_type(8)))  float        v8f;
typedef __attribute__((ext_vector_type(4)))  float        f32x4;
typedef __attribute__((ext_vector_type(2)))  unsigned int u32x2;

#define BLK_M 128
#define BLK_N 128
#define BLK_K 32
#define LDS_STRIDE 36   // 32 + 4 halves pad

// Build M (196x196, f16) from qk (27x27 f32).
__global__ void build_mix_matrix(const float* __restrict__ qk,
                                 _Float16* __restrict__ M) {
  int idx = blockIdx.x * blockDim.x + threadIdx.x;
  if (idx >= 196 * 196) return;
  int p = idx / 196, q = idx % 196;
  int py = p / 14, px = p % 14;
  int qy = q / 14, qx = q % 14;
  // cross-correlation, padding 13: out[p] += in[q] * qk[13+qy-py, 13+qx-px]
  float w = qk[(13 + qy - py) * 27 + (13 + qx - px)];
  M[idx] = (_Float16)w;
}

// Load a 16x32 f16 A fragment from LDS (row-major, stride LDS_STRIDE halves).
// Layout: lanes 0-15 row m=lane, K=0..15; lanes 16-31 same rows, K=16..31.
// VGPR j -> K=2j,2j+1 (+kh).
__device__ inline v16h load_frag_a(const _Float16* s, int rowBase, int lane) {
  union { v16h v; unsigned int u[8]; } f;
  int row = rowBase + (lane & 15);
  int kh  = ((lane >> 4) & 1) * 16;
  const unsigned short* p = (const unsigned short*)s + row * LDS_STRIDE;
#pragma unroll
  for (int j = 0; j < 8; ++j) {
    int k = 2 * j + kh;
    f.u[j] = *(const unsigned int*)(p + k);
  }
  return f.v;
}

// Load a 32x16 f16 B fragment from LDS stored N-major (stride LDS_STRIDE).
// Layout: lanes 0-15 col n=lane, K=0..15 (VGPR j -> K=2j,2j+1);
// lanes 16-31 same cols, K=16..31.
__device__ inline v16h load_frag_b(const _Float16* s, int colBase, int lane) {
  union { v16h v; unsigned int u[8]; } f;
  int col = colBase + (lane & 15);
  int kh  = ((lane >> 4) & 1) * 16;
  const unsigned short* p = (const unsigned short*)s + col * LDS_STRIDE + kh;
#pragma unroll
  for (int j = 0; j < 8; ++j) f.u[j] = *(const unsigned int*)(p + 2 * j);
  return f.v;
}

// Direct (load->convert->store) staging of one A tile into LDS.
template <bool AF32>
__device__ inline void stage_A_direct(_Float16* dst, const void* A, int m0,
                                      int k0, int Mdim, int Kdim, int lda,
                                      int tid) {
#pragma unroll
  for (int it = 0; it < 4; ++it) {
    int idx = tid + it * 256;        // 1024 groups of 4 halves
    int r = idx >> 3, kk = (idx & 7) * 4;
    int gr = m0 + r, gk = k0 + kk;
    bool ok = (gr < Mdim) && (gk < Kdim);
    _Float16 h4[4];
    if constexpr (AF32) {
      f32x4 a4 = ok ? *(const f32x4*)((const float*)A + (long)gr * lda + gk)
                    : (f32x4)0.0f;
      h4[0] = (_Float16)a4.x; h4[1] = (_Float16)a4.y;
      h4[2] = (_Float16)a4.z; h4[3] = (_Float16)a4.w;
    } else {
      u32x2 a2 = ok ? *(const u32x2*)((const _Float16*)A + (long)gr * lda + gk)
                    : (u32x2)0u;
      *(u32x2*)h4 = a2;
    }
    *(u32x2*)&dst[r * LDS_STRIDE + kk] = *(const u32x2*)h4;
  }
}

// Direct staging of one B tile (K x 128, transposed to N-major) into LDS.
template <bool BF32>
__device__ inline void stage_B_direct(_Float16* dst, const void* B, int n0,
                                      int k0, int Kdim, int Ndim, int ldb,
                                      int tid) {
  int n = tid & 127;
  int kh = (tid >> 7) * 16;
  int gn = n0 + n;
  _Float16 tmp[16];
#pragma unroll
  for (int j = 0; j < 16; ++j) {
    int gk = k0 + kh + j;
    bool ok = (gk < Kdim) && (gn < Ndim);
    if constexpr (BF32)
      tmp[j] = (_Float16)(ok ? ((const float*)B)[(long)gk * ldb + gn] : 0.0f);
    else
      tmp[j] = ok ? ((const _Float16*)B)[(long)gk * ldb + gn] : (_Float16)0.0f;
  }
#pragma unroll
  for (int j = 0; j < 4; ++j)
    *(u32x2*)&dst[n * LDS_STRIDE + kh + 4 * j] = *(const u32x2*)&tmp[4 * j];
}

// Generic tiled WMMA GEMM:  C = A(MxK) * B(KxN) [+ bias], batched via grid.z.
// AF32/BF32: source is f32 (converted on staging), else f16. CF32: f32 out.
template <bool AF32, bool BF32, bool CF32>
__global__ __launch_bounds__(256) void gemm_wmma_kernel(
    const void* __restrict__ Av, long strideA,
    const void* __restrict__ Bv, long strideB,
    void* __restrict__ Cv, long strideC,
    int Mdim, int Kdim, int Ndim,
    int lda, int ldb, int ldc,
    const float* __restrict__ bias) {
  __shared__ _Float16 As[2][BLK_M * LDS_STRIDE];
  __shared__ _Float16 Bs[2][BLK_N * LDS_STRIDE];

  const int tid   = threadIdx.x;
  const int lane  = tid & 31;
  const int wave  = tid >> 5;          // 8 waves
  const int waveM = wave >> 2;         // 0..1 -> 64 rows each
  const int waveN = wave & 3;          // 0..3 -> 32 cols each

  const int m0 = blockIdx.x * BLK_M;
  const int n0 = blockIdx.y * BLK_N;
  const long z = blockIdx.z;

  const void* A = (const char*)Av + (AF32 ? 4l : 2l) * z * strideA;
  const void* B = (const char*)Bv + (BF32 ? 4l : 2l) * z * strideB;
  void*       C = (char*)Cv + (CF32 ? 4l : 2l) * z * strideC;

  v8f acc[4][2];
#pragma unroll
  for (int i = 0; i < 4; ++i)
#pragma unroll
    for (int j = 0; j < 2; ++j) acc[i][j] = (v8f)0.0f;

  const int nsteps = (Kdim + BLK_K - 1) / BLK_K;

  // Prologue: stage K-step 0 into buffer 0.
  stage_A_direct<AF32>(&As[0][0], A, m0, 0, Mdim, Kdim, lda, tid);
  stage_B_direct<BF32>(&Bs[0][0], B, n0, 0, Kdim, Ndim, ldb, tid);
  __syncthreads();

  for (int s = 0; s < nsteps; ++s) {
    const int cur = s & 1, nxt = cur ^ 1;
    const int k1 = (s + 1) * BLK_K;
    const bool haveNext = (s + 1) < nsteps;

    // ---- issue next-tile global traffic (overlaps with compute below) ----
    f32x4    aF[4];
    u32x2    aH[4];
    _Float16 bT[16];
    bool aAsync = false;
    if (haveNext) {
      if constexpr (!AF32) {
        if ((m0 + BLK_M <= Mdim) && (k1 + BLK_K <= Kdim)) {
          // f16 A tile fully in-bounds: async DMA straight into LDS[nxt].
          aAsync = true;
#pragma unroll
          for (int it = 0; it < 4; ++it) {
            int idx = tid + it * 256;
            int r = idx >> 3, kk = (idx & 7) * 4;
            const void* g = (const _Float16*)A + (long)(m0 + r) * lda + k1 + kk;
            unsigned lofs =
                (unsigned)(uintptr_t)&As[nxt][r * LDS_STRIDE + kk];
            asm volatile("global_load_async_to_lds_b64 %0, %1, off"
                         :: "v"(lofs), "v"(g) : "memory");
          }
        } else {
#pragma unroll
          for (int it = 0; it < 4; ++it) {
            int idx = tid + it * 256;
            int r = idx >> 3, kk = (idx & 7) * 4;
            int gr = m0 + r, gk = k1 + kk;
            bool ok = (gr < Mdim) && (gk < Kdim);
            aH[it] = ok
                ? *(const u32x2*)((const _Float16*)A + (long)gr * lda + gk)
                : (u32x2)0u;
          }
        }
      } else {
#pragma unroll
        for (int it = 0; it < 4; ++it) {
          int idx = tid + it * 256;
          int r = idx >> 3, kk = (idx & 7) * 4;
          int gr = m0 + r, gk = k1 + kk;
          bool ok = (gr < Mdim) && (gk < Kdim);
          aF[it] = ok ? *(const f32x4*)((const float*)A + (long)gr * lda + gk)
                      : (f32x4)0.0f;
        }
      }
      {
        int n = tid & 127, kh = (tid >> 7) * 16, gn = n0 + n;
#pragma unroll
        for (int j = 0; j < 16; ++j) {
          int gk = k1 + kh + j;
          bool ok = (gk < Kdim) && (gn < Ndim);
          if constexpr (BF32)
            bT[j] = (_Float16)(ok ? ((const float*)B)[(long)gk * ldb + gn]
                                  : 0.0f);
          else
            bT[j] = ok ? ((const _Float16*)B)[(long)gk * ldb + gn]
                       : (_Float16)0.0f;
        }
      }
      if ((s + 2) < nsteps) {  // warm L2/WGP$ two steps ahead
        int k2 = (s + 2) * BLK_K;
        const char* Ab = (const char*)A;
        const char* Bb = (const char*)B;
        __builtin_prefetch(
            Ab + ((long)(m0 + (tid >> 1)) * lda + k2) * (AF32 ? 4 : 2), 0, 0);
        __builtin_prefetch(
            Bb + ((long)(k2 + (tid >> 7) * 16) * ldb + n0 + (tid & 127)) *
                     (BF32 ? 4 : 2), 0, 0);
      }
    }

    // ---- compute current tile: 4x2 fragments of v_wmma_f32_16x16x32_f16 ---
    v16h af[4], bf2[2];
#pragma unroll
    for (int i = 0; i < 4; ++i)
      af[i] = load_frag_a(&As[cur][0], waveM * 64 + i * 16, lane);
#pragma unroll
    for (int j = 0; j < 2; ++j)
      bf2[j] = load_frag_b(&Bs[cur][0], waveN * 32 + j * 16, lane);
#pragma unroll
    for (int i = 0; i < 4; ++i)
#pragma unroll
      for (int j = 0; j < 2; ++j)
        acc[i][j] = __builtin_amdgcn_wmma_f32_16x16x32_f16(
            false, af[i], false, bf2[j], (short)0, acc[i][j], false, false);

    // ---- drain staged registers into LDS[nxt] ----
    if (haveNext) {
      if constexpr (AF32) {
#pragma unroll
        for (int it = 0; it < 4; ++it) {
          int idx = tid + it * 256;
          int r = idx >> 3, kk = (idx & 7) * 4;
          _Float16 h4[4];
          h4[0] = (_Float16)aF[it].x; h4[1] = (_Float16)aF[it].y;
          h4[2] = (_Float16)aF[it].z; h4[3] = (_Float16)aF[it].w;
          *(u32x2*)&As[nxt][r * LDS_STRIDE + kk] = *(const u32x2*)h4;
        }
      } else if (!aAsync) {
#pragma unroll
        for (int it = 0; it < 4; ++it) {
          int idx = tid + it * 256;
          int r = idx >> 3, kk = (idx & 7) * 4;
          *(u32x2*)&As[nxt][r * LDS_STRIDE + kk] = aH[it];
        }
      }
      int n = tid & 127, kh = (tid >> 7) * 16;
#pragma unroll
      for (int j = 0; j < 4; ++j)
        *(u32x2*)&Bs[nxt][n * LDS_STRIDE + kh + 4 * j] =
            *(const u32x2*)&bT[4 * j];
    }
    if (aAsync) asm volatile("s_wait_asynccnt 0x0" ::: "memory");
    __syncthreads();
  }

  // ---- epilogue: C/D layout -> lanes 0-15 N=lane M=i, lanes 16-31 M=8+i ---
  const int mBase = m0 + waveM * 64 + ((lane >> 4) & 1) * 8;
  const int nBase = n0 + waveN * 32 + (lane & 15);
#pragma unroll
  for (int i = 0; i < 4; ++i) {
#pragma unroll
    for (int j = 0; j < 2; ++j) {
      int n = nBase + j * 16;
      if (n >= Ndim) continue;
      float bv = bias ? bias[n] : 0.0f;
#pragma unroll
      for (int e = 0; e < 8; ++e) {
        int m = mBase + i * 16 + e;
        if (m < Mdim) {
          float val = acc[i][j][e] + bv;
          if constexpr (CF32)
            ((float*)C)[(long)m * ldc + n] = val;
          else
            ((_Float16*)C)[(long)m * ldc + n] = (_Float16)val;
        }
      }
    }
  }
}

extern "C" void kernel_launch(void* const* d_in, const int* in_sizes, int n_in,
                              void* d_out, int out_size, void* d_ws, size_t ws_size,
                              hipStream_t stream) {
  const float* x  = (const float*)d_in[0];  // 128*196*384
  const float* Wv = (const float*)d_in[1];  // 384*512
  const float* qk = (const float*)d_in[2];  // 1*1*27*27
  const float* Wo = (const float*)d_in[3];  // 512*384
  const float* bo = (const float*)d_in[4];  // 384
  float* out = (float*)d_out;               // 128*196*384

  const int B = 128, N = 196, D = 384, E = 512;
  const int ROWS = B * N;                   // 25088

  _Float16* wsM   = (_Float16*)d_ws;              // 196*196
  _Float16* wsV   = wsM + 196 * 196;              // 25088*512 (16B-aligned)
  _Float16* wsMid = wsV + (long)ROWS * E;         // 25088*512

  // 1) Materialize the 196x196 mixing matrix from qk.
  build_mix_matrix<<<(196 * 196 + 255) / 256, 256, 0, stream>>>(qk, wsM);

  // 2) v = x @ Wv   (f32 in -> f16 out)
  gemm_wmma_kernel<true, true, false>
      <<<dim3(ROWS / BLK_M, E / BLK_N, 1), 256, 0, stream>>>(
          x, 0, Wv, 0, wsV, 0, ROWS, D, E, D, E, E, nullptr);

  // 3) mid[b] = M @ v[b]   (batched over 128)
  gemm_wmma_kernel<false, false, false>
      <<<dim3((N + BLK_M - 1) / BLK_M, E / BLK_N, B), 256, 0, stream>>>(
          wsM, 0, wsV, (long)N * E, wsMid, (long)N * E,
          N, N, E, N, E, E, nullptr);

  // 4) out = mid @ Wo + bo   (f16 in -> f32 out)
  gemm_wmma_kernel<false, true, true>
      <<<dim3(ROWS / BLK_M, D / BLK_N, 1), 256, 0, stream>>>(
          wsMid, 0, Wo, 0, out, 0, ROWS, E, D, E, D, D, bo);
}
